// AWD_LSTM_6451040878885
// MI455X (gfx1250) — matrix operand order repaired and checked by reading, not compiled, not run on hardware
//
#include <hip/hip_runtime.h>
#include <stdint.h>

// ---------------- problem dimensions ----------------
#define T_SEQ 70
#define BATCH 80
#define EMBD  400
#define HID   1150
#define NTOK  33278
#define H4    4600          // 4*HID
// padded dims: K padded to mult of 32 (WMMA K), N padded to mult of 16
#define EP    416           // EMBD -> 416
#define HP    1152          // HID  -> 1152
#define H4P   4608          // 4H   -> 4608
#define VP    33280         // NTOK -> 33280

#define LDS_STRIDE 40       // 32 k-elems + 8 pad (80B rows -> conflict-free b128 reads)
#define KSTEPS (HP / 32)    // 36

typedef unsigned short u16;
typedef __attribute__((ext_vector_type(16))) __bf16 v16bf;
typedef __attribute__((ext_vector_type(8)))  float  v8f;

// matches the builtin prototype "int __vector(4)" (vector_size style)
typedef int v4i __attribute__((vector_size(16)));
typedef __attribute__((address_space(1))) v4i* gv4i_p;   // global int4*
typedef __attribute__((address_space(3))) v4i* lv4i_p;   // LDS int4*

union Frag { uint4 q[2]; v16bf v; };

#if __has_builtin(__builtin_amdgcn_global_load_async_to_lds_b128)
#define ASYNC_LDS_OK 1
#else
#define ASYNC_LDS_OK 0
#endif

__device__ __forceinline__ u16 f2bf(float f) {
  unsigned u = __float_as_uint(f);
  u += 0x7FFFu + ((u >> 16) & 1u);   // round-to-nearest-even
  return (u16)(u >> 16);
}

__device__ __forceinline__ float sigmoidf_(float x) {
  return 1.0f / (1.0f + expf(-x));
}

// ---------------------------------------------------------------------------
// Multi-M K-loop: one B fragment feeds all 5 M-tiles (M=80) of the step GEMM.
// A: [80, kpad] bf16 row-major, B: [N, kpad] bf16 row-major (both zero padded).
// Fragment layout used here (wave32; contiguous k, not the ISA table order):
//   A (16x32): lane L -> row M=L%16; K = (L/16)*16..+16 (contiguous)
//   B (32x16): lane L -> col N=L%16; K = (L/16)*16..+16 (contiguous)
// ---------------------------------------------------------------------------
__device__ __forceinline__ void gemm5(v8f acc[5],
    const u16* __restrict__ A, const u16* __restrict__ B,
    int kpad, int brow, int lane)
{
  const int half = lane >> 4;
  const int l16  = lane & 15;
  const u16* bp = B + (size_t)(brow + l16) * kpad;
  for (int k = 0; k < kpad; k += 32) {
    Frag fb;
    const u16* bk = bp + k + half * 16;
    fb.q[0] = *(const uint4*)(bk);
    fb.q[1] = *(const uint4*)(bk + 8);
    __builtin_prefetch(bp + k + 32, 0, 1);
    #pragma unroll
    for (int mi = 0; mi < 5; ++mi) {
      Frag fa;
      const u16* ar = A + (size_t)(mi * 16 + l16) * kpad + k + half * 16;
      fa.q[0] = *(const uint4*)(ar);
      fa.q[1] = *(const uint4*)(ar + 8);
      acc[mi] = __builtin_amdgcn_wmma_f32_16x16x32_bf16(false, fa.v, false, fb.v,
                                                        (short)0, acc[mi], false, false);
    }
  }
}

// ---------------------------------------------------------------------------
// pre = X*Wx^T + H*Wh^T + b_ih + b_hh  ->  [BATCH, H4P] fp32
// one wave per N-tile, all 5 M-tiles register-blocked (weights streamed 1x).
// block (32,4), grid (288/4 = 72)
// ---------------------------------------------------------------------------
__global__ void lstm_pre_gemm(const u16* __restrict__ X, int kx,
                              const u16* __restrict__ Wx,
                              const u16* __restrict__ Hm,
                              const u16* __restrict__ Wh,
                              const float* __restrict__ bx,
                              const float* __restrict__ bh,
                              float* __restrict__ out)
{
  const int lane  = threadIdx.x;
  const int ntile = blockIdx.x * blockDim.y + threadIdx.y;  // 0..287
  v8f acc[5];
  #pragma unroll
  for (int mi = 0; mi < 5; ++mi)
    acc[mi] = (v8f){0.f,0.f,0.f,0.f,0.f,0.f,0.f,0.f};

  gemm5(acc, X,  Wx, kx, ntile * 16, lane);
  gemm5(acc, Hm, Wh, HP, ntile * 16, lane);

  const int half = lane >> 4, l16 = lane & 15;
  const int col  = ntile * 16 + l16;
  const float bias = (col < H4) ? (bx[col] + bh[col]) : 0.0f;
  #pragma unroll
  for (int mi = 0; mi < 5; ++mi) {
    const int rbase = mi * 16 + half * 8;   // D: VGPR r -> row half*8+r, col lane%16
    #pragma unroll
    for (int r = 0; r < 8; ++r)
      out[(size_t)(rbase + r) * H4P + col] = acc[mi][r] + bias;
  }
}

// ---------------------------------------------------------------------------
// gates i|f|o|g -> update c,h; emit h as fp32 state + bf16 (padded) operand
// ---------------------------------------------------------------------------
__global__ void lstm_cell_ew(const float* __restrict__ pre,
                             float* __restrict__ cst,
                             float* __restrict__ hst,
                             u16*  __restrict__ hpad,
                             u16*  __restrict__ outsb)
{
  int idx = blockIdx.x * blockDim.x + threadIdx.x;
  if (idx >= BATCH * HID) return;
  int b = idx / HID, k = idx - b * HID;
  const float* p = pre + (size_t)b * H4P;
  float i = sigmoidf_(p[k]);
  float f = sigmoidf_(p[HID + k]);
  float o = sigmoidf_(p[2 * HID + k]);
  float g = tanhf(p[3 * HID + k]);
  float c = f * cst[idx] + i * g;
  float h = o * tanhf(c);
  cst[idx] = c;
  hst[idx] = h;
  u16 hb = f2bf(h);
  hpad[(size_t)b * HP + k] = hb;
  if (outsb) outsb[(size_t)b * HP + k] = hb;
}

// ---------------------------------------------------------------------------
// Async global->LDS staging of one A chunk [80 x 32] (ASYNCcnt path, gfx1250)
// ---------------------------------------------------------------------------
#if ASYNC_LDS_OK
__device__ __forceinline__ void stage_async(const u16* A, u16* dst,
                                            int mrow0, int k, int tid)
{
  for (int e = tid; e < 320; e += 128) {
    int row = e >> 2, kq = e & 3;
    __builtin_amdgcn_global_load_async_to_lds_b128(
        (gv4i_p)(A + (size_t)(mrow0 + row) * HP + k + kq * 8),
        (lv4i_p)(dst + row * LDS_STRIDE + kq * 8),
        0, 0);
  }
}
#endif

__device__ __forceinline__ void async_wait0()
{
#if __has_builtin(__builtin_amdgcn_s_wait_asynccnt)
  __builtin_amdgcn_s_wait_asynccnt(0);
#else
  asm volatile("s_wait_asynccnt 0" ::: "memory");
#endif
}

// one decoder K-step: 2 B fragments (global) x 5 A fragments (LDS) = 10 WMMA
__device__ __forceinline__ void dec_step(v8f acc[5][2], const u16* buf,
    const u16* b0, const u16* b1, int k, int half, int l16)
{
  Frag fb0, fb1;
  const u16* p0 = b0 + k + half * 16;
  fb0.q[0] = *(const uint4*)(p0);
  fb0.q[1] = *(const uint4*)(p0 + 8);
  const u16* p1 = b1 + k + half * 16;
  fb1.q[0] = *(const uint4*)(p1);
  fb1.q[1] = *(const uint4*)(p1 + 8);
  __builtin_prefetch(b0 + k + 32, 0, 1);
  __builtin_prefetch(b1 + k + 32, 0, 1);
  #pragma unroll
  for (int mi = 0; mi < 5; ++mi) {
    Frag fa;
    const u16* ar = buf + (mi * 16 + l16) * LDS_STRIDE + half * 16;
    fa.q[0] = *(const uint4*)(ar);
    fa.q[1] = *(const uint4*)(ar + 8);
    acc[mi][0] = __builtin_amdgcn_wmma_f32_16x16x32_bf16(false, fa.v, false, fb0.v,
                                                         (short)0, acc[mi][0], false, false);
    acc[mi][1] = __builtin_amdgcn_wmma_f32_16x16x32_bf16(false, fa.v, false, fb1.v,
                                                         (short)0, acc[mi][1], false, false);
  }
}

// ---------------------------------------------------------------------------
// decoded[m, v] = outs[m, :] . w_dec[v, :] + b_dec[v]
// Workgroup owns an 80-row M-block; A K-chunks double-buffered in LDS via
// async global->LDS copies (shared by 4 waves). Each wave: 2 N-tiles x
// 5 M-tiles = 10 accumulators; B fragments reused 5x, A (LDS) reused 2x.
// block (32,4), grid (2080/2/4 = 260, 5600/80 = 70)
// ---------------------------------------------------------------------------
__global__ void decoder_gemm(const u16* __restrict__ A,   // [5600, HP]
                             const u16* __restrict__ W,   // [VP, HP]
                             const float* __restrict__ bias,
                             float* __restrict__ out)     // [5600, NTOK]
{
  const int lane  = threadIdx.x;
  const int ty    = threadIdx.y;
  const int tid   = ty * 32 + lane;
  const int pair  = blockIdx.x * blockDim.y + ty;   // 0..1039 (pairs of N-tiles)
  const int mrow0 = blockIdx.y * 80;
  const int half  = lane >> 4, l16 = lane & 15;

  v8f acc[5][2];
  #pragma unroll
  for (int mi = 0; mi < 5; ++mi)
    #pragma unroll
    for (int ni = 0; ni < 2; ++ni)
      acc[mi][ni] = (v8f){0.f,0.f,0.f,0.f,0.f,0.f,0.f,0.f};

  const u16* b0 = W + (size_t)(pair * 32 + l16) * HP;       // B tile 0 row
  const u16* b1 = W + (size_t)(pair * 32 + 16 + l16) * HP;  // B tile 1 row

#if ASYNC_LDS_OK
  __shared__ __align__(16) u16 As[2][80 * LDS_STRIDE];
  stage_async(A, &As[0][0], mrow0, 0, tid);          // kick off chunk 0
  for (int kb = 0; kb < KSTEPS; ++kb) {
    const int k = kb * 32;
    async_wait0();                                   // own async ops landed
    __syncthreads();                                 // -> all waves' staging visible
    if (kb + 1 < KSTEPS)                             // overlap next copy w/ compute
      stage_async(A, &As[(kb + 1) & 1][0], mrow0, k + 32, tid);
    dec_step(acc, &As[kb & 1][0], b0, b1, k, half, l16);
  }
#else
  __shared__ __align__(16) u16 As[1][80 * LDS_STRIDE];
  for (int k = 0; k < HP; k += 32) {
    __syncthreads();
    #pragma unroll 1
    for (int e = tid; e < 320; e += 128) {           // synchronous staging fallback
      int row = e >> 2, kq = e & 3;
      *(uint4*)(&As[0][0] + row * LDS_STRIDE + kq * 8) =
          *(const uint4*)(A + (size_t)(mrow0 + row) * HP + k + kq * 8);
    }
    __syncthreads();
    dec_step(acc, &As[0][0], b0, b1, k, half, l16);
  }
#endif

  #pragma unroll
  for (int ni = 0; ni < 2; ++ni) {
    const int col = pair * 32 + ni * 16 + l16;
    if (col >= NTOK) continue;              // padded N tail
    const float b = bias[col];
    #pragma unroll
    for (int mi = 0; mi < 5; ++mi) {
      const int rbase = mrow0 + mi * 16 + half * 8;
      #pragma unroll
      for (int r = 0; r < 8; ++r)
        out[(size_t)(rbase + r) * NTOK + col] = acc[mi][ni][r] + b;
    }
  }
}

// ---------------------------------------------------------------------------
// fp32 [R,K] -> bf16 [RP,KP], zero padded
// ---------------------------------------------------------------------------
__global__ void cvt_pad_bf16(const float* __restrict__ src, u16* __restrict__ dst,
                             int R, int K, int KP_, long long total)
{
  long long idx = (long long)blockIdx.x * blockDim.x + threadIdx.x;
  if (idx >= total) return;
  int kp = (int)(idx % KP_);
  long long r = idx / KP_;
  float v = (r < R && kp < K) ? src[r * (long long)K + kp] : 0.0f;
  dst[idx] = f2bf(v);
}

__global__ void embed_gather(const int* __restrict__ x, const float* __restrict__ emb,
                             u16* __restrict__ xe)
{
  long long idx = (long long)blockIdx.x * blockDim.x + threadIdx.x;
  const long long total = (long long)T_SEQ * BATCH * EP;
  if (idx >= total) return;
  int col = (int)(idx % EP);
  long long m = idx / EP;
  int tok = x[m];
  float v = (col < EMBD) ? emb[(size_t)tok * EMBD + col] : 0.0f;
  xe[idx] = f2bf(v);
}

__global__ void init_state(const float* __restrict__ h0, const float* __restrict__ c0,
                           float* __restrict__ hst, float* __restrict__ cst,
                           u16* __restrict__ hpad)
{
  int idx = blockIdx.x * blockDim.x + threadIdx.x;   // over 3*B*HP
  if (idx >= 3 * BATCH * HP) return;
  int k  = idx % HP;
  int lb = idx / HP;                                 // l*B + b
  if (k < HID) {
    float h = h0[(size_t)lb * HID + k];
    hst[(size_t)lb * HID + k] = h;
    cst[(size_t)lb * HID + k] = c0[(size_t)lb * HID + k];
    hpad[idx] = f2bf(h);
  } else {
    hpad[idx] = 0;
  }
}

__global__ void copy_states_out(const float* __restrict__ hst,
                                const float* __restrict__ cst,
                                float* __restrict__ out)
{
  int idx = blockIdx.x * blockDim.x + threadIdx.x;
  const int n = 3 * BATCH * HID;
  if (idx >= 2 * n) return;
  out[idx] = (idx < n) ? hst[idx] : cst[idx - n];
}

// ---------------------------------------------------------------------------
extern "C" void kernel_launch(void* const* d_in, const int* in_sizes, int n_in,
                              void* d_out, int out_size, void* d_ws, size_t ws_size,
                              hipStream_t stream)
{
  (void)in_sizes; (void)n_in; (void)out_size; (void)ws_size;
  const int*   x     = (const int*)  d_in[0];
  const float* h0    = (const float*)d_in[1];
  const float* c0    = (const float*)d_in[2];
  const float* emb   = (const float*)d_in[3];
  const float* w_ih0 = (const float*)d_in[4];
  const float* b_ih0 = (const float*)d_in[5];
  const float* w_hh0 = (const float*)d_in[6];
  const float* b_hh0 = (const float*)d_in[7];
  const float* w_ih1 = (const float*)d_in[8];
  const float* b_ih1 = (const float*)d_in[9];
  const float* w_hh1 = (const float*)d_in[10];
  const float* b_hh1 = (const float*)d_in[11];
  const float* w_ih2 = (const float*)d_in[12];
  const float* b_ih2 = (const float*)d_in[13];
  const float* w_hh2 = (const float*)d_in[14];
  const float* b_hh2 = (const float*)d_in[15];
  const float* w_dec = (const float*)d_in[16];
  const float* b_dec = (const float*)d_in[17];
  float* out = (float*)d_out;

  // ---- workspace carve-out (256B aligned slabs) ----
  char* base = (char*)d_ws;
  size_t off = 0;
  auto alloc = [&](size_t bytes) -> char* {
    char* p = base + off;
    off = (off + bytes + 255) & ~(size_t)255;
    return p;
  };
  u16* wih0  = (u16*)alloc((size_t)H4P * EP * 2);
  u16* whh0  = (u16*)alloc((size_t)H4P * HP * 2);
  u16* wih1  = (u16*)alloc((size_t)H4P * HP * 2);
  u16* whh1  = (u16*)alloc((size_t)H4P * HP * 2);
  u16* wih2  = (u16*)alloc((size_t)H4P * HP * 2);
  u16* whh2  = (u16*)alloc((size_t)H4P * HP * 2);
  u16* wdec  = (u16*)alloc((size_t)VP  * HP * 2);
  u16* xe    = (u16*)alloc((size_t)T_SEQ * BATCH * EP * 2);
  u16* hpad  = (u16*)alloc((size_t)3 * BATCH * HP * 2);
  u16* outsb = (u16*)alloc((size_t)T_SEQ * BATCH * HP * 2);
  float* pre = (float*)alloc((size_t)BATCH * H4P * 4);
  float* hst = (float*)alloc((size_t)3 * BATCH * HID * 4);
  float* cst = (float*)alloc((size_t)3 * BATCH * HID * 4);

  // ---- 1. weight conversion to padded bf16 ----
  auto cvt = [&](const float* s, u16* d, int R, int K, int RP, int KP_) {
    long long total = (long long)RP * KP_;
    cvt_pad_bf16<<<(int)((total + 255) / 256), 256, 0, stream>>>(s, d, R, K, KP_, total);
  };
  cvt(w_ih0, wih0, H4, EMBD, H4P, EP);
  cvt(w_hh0, whh0, H4, HID,  H4P, HP);
  cvt(w_ih1, wih1, H4, HID,  H4P, HP);
  cvt(w_hh1, whh1, H4, HID,  H4P, HP);
  cvt(w_ih2, wih2, H4, HID,  H4P, HP);
  cvt(w_hh2, whh2, H4, HID,  H4P, HP);
  cvt(w_dec, wdec, NTOK, HID, VP, HP);

  // ---- 2. embedding gather + bf16 ----
  {
    long long total = (long long)T_SEQ * BATCH * EP;
    embed_gather<<<(int)((total + 255) / 256), 256, 0, stream>>>(x, emb, xe);
  }
  // ---- 3. state init; zero the decoder A-operand (pad cols) ----
  init_state<<<(3 * BATCH * HP + 255) / 256, 256, 0, stream>>>(h0, c0, hst, cst, hpad);
  (void)hipMemsetAsync(outsb, 0, (size_t)T_SEQ * BATCH * HP * 2, stream);

  // ---- 4. sequential LSTM over T (stream order = recurrence order) ----
  dim3 blkG(32, 4);
  dim3 grdG(H4P / 16 / 4);                // 72 blocks; 288 waves; 5 M-tiles/wave
  const int ewBlocks = (BATCH * HID + 255) / 256;

  u16* hp0 = hpad;
  u16* hp1 = hpad + (size_t)BATCH * HP;
  u16* hp2 = hpad + (size_t)2 * BATCH * HP;
  float* c0s = cst;
  float* c1s = cst + (size_t)BATCH * HID;
  float* c2s = cst + (size_t)2 * BATCH * HID;
  float* h0s = hst;
  float* h1s = hst + (size_t)BATCH * HID;
  float* h2s = hst + (size_t)2 * BATCH * HID;

  for (int t = 0; t < T_SEQ; ++t) {
    const u16* xt = xe + (size_t)t * BATCH * EP;
    lstm_pre_gemm<<<grdG, blkG, 0, stream>>>(xt,  EP, wih0, hp0, whh0, b_ih0, b_hh0, pre);
    lstm_cell_ew<<<ewBlocks, 256, 0, stream>>>(pre, c0s, h0s, hp0, (u16*)nullptr);
    lstm_pre_gemm<<<grdG, blkG, 0, stream>>>(hp0, HP, wih1, hp1, whh1, b_ih1, b_hh1, pre);
    lstm_cell_ew<<<ewBlocks, 256, 0, stream>>>(pre, c1s, h1s, hp1, (u16*)nullptr);
    lstm_pre_gemm<<<grdG, blkG, 0, stream>>>(hp1, HP, wih2, hp2, whh2, b_ih2, b_hh2, pre);
    lstm_cell_ew<<<ewBlocks, 256, 0, stream>>>(pre, c2s, h2s, hp2,
                                               outsb + (size_t)t * BATCH * HP);
  }

  // ---- 5. decoder GEMM: async-LDS double-buffered A, 5x2 register tiles ----
  dim3 blkD(32, 4);
  dim3 grdD(VP / 32 / 4, (T_SEQ * BATCH) / 80);   // (260, 70)
  decoder_gemm<<<grdD, blkD, 0, stream>>>(outsb, wdec, b_dec, out);

  // ---- 6. final h/c states into output tail ----
  copy_states_out<<<(2 * 3 * BATCH * HID + 255) / 256, 256, 0, stream>>>(
      hst, cst, out + (size_t)T_SEQ * BATCH * NTOK);
}
